// CrossLayer_21964462752441
// MI455X (gfx1250) — compile-verified, matrix operand order checked by reading, not run on hardware
//
#include <hip/hip_runtime.h>
#include <hip/hip_bf16.h>

// ---------------------------------------------------------------------------
// CDNA5 (gfx1250) WMMA types & helpers
// ---------------------------------------------------------------------------
typedef __attribute__((ext_vector_type(16))) __bf16 v16bf;
typedef __attribute__((ext_vector_type(8)))  float  v8f;

static constexpr int Bn  = 32;    // batch
static constexpr int LU  = 128;   // cur utt length
static constexpr int LC  = 512;   // context length
static constexpr int Sn  = 30;    // slots
static constexpr int Dm  = 768;   // model dim
static constexpr int Hh  = 12;    // heads
static constexpr int DK  = 64;    // head dim
static constexpr int DFF = 1152;

__device__ __forceinline__ unsigned short f2bf(float f) {
    unsigned int u = __float_as_uint(f);
    u += 0x7FFFu + ((u >> 16) & 1u);           // round-to-nearest-even
    return (unsigned short)(u >> 16);
}
__device__ __forceinline__ unsigned int packbf(float lo, float hi) {
    return (unsigned int)f2bf(lo) | ((unsigned int)f2bf(hi) << 16);
}

// Load a 16-element bf16 fragment (one lane's share of A [16xK-major] or of
// B stored K-contiguous per column). `p` points at (row, kbase) with the K
// dimension contiguous in ushorts. Element i covers K = (i<4?0:16)+2*(i&3),
// so the 8 dword reads merge into two b128 LDS loads.
__device__ __forceinline__ v16bf load_frag(const unsigned short* p) {
    union { unsigned int u[8]; v16bf bf; } f;
#pragma unroll
    for (int i = 0; i < 8; ++i) {
        int k = ((i & 4) << 2) + ((i & 3) << 1);
        f.u[i] = *(const unsigned int*)(p + k);
    }
    return f.bf;
}

__device__ __forceinline__ v8f wmma_bf16(v16bf a, v16bf b, v8f c) {
    return __builtin_amdgcn_wmma_f32_16x16x32_bf16(
        false, a, false, b, (short)0, c, false, false);
}

__device__ __forceinline__ float redmax16(float v) {
#pragma unroll
    for (int m = 1; m < 16; m <<= 1) v = fmaxf(v, __shfl_xor(v, m, 32));
    return v;
}
__device__ __forceinline__ float redsum16(float v) {
#pragma unroll
    for (int m = 1; m < 16; m <<= 1) v += __shfl_xor(v, m, 32);
    return v;
}

// ---------------------------------------------------------------------------
// GEMM: C[M,N] = A[M,K] @ W[K,N] + bias[N] (+resid[M,N]) (ReLU?)
// fp32 in/out, bf16 WMMA compute. 64x64x32 tiles, 4 waves / 128 threads,
// each wave owns a 16x64 strip (4 WMMAs per K-step).
// REQUIRES: M % 64 == 0, N % 64 == 0, K % 32 == 0 (true for all call sites).
// RELU / RESID are compile-time so the epilogue is branch-free.
// ---------------------------------------------------------------------------
template <bool RELU, bool RESID>
__global__ void __launch_bounds__(128)
gemm_kernel(const float* __restrict__ A, const float* __restrict__ W,
            const float* __restrict__ bias, const float* __restrict__ resid,
            float* __restrict__ C, int M, int N, int K)
{
    __shared__ __align__(16) unsigned int sA [64 * 16];  // [row][kpair]
    __shared__ __align__(16) unsigned int sBt[64 * 16];  // [col][kpair] (W^T)

    const int tid  = threadIdx.x;
    const int lane = tid & 31;
    const int wave = tid >> 5;              // row tile 0..3
    const int l16  = lane & 15;
    const int hi   = lane >> 4;
    const int mBase = blockIdx.y * 64;
    const int nBase = blockIdx.x * 64;

    v8f acc0 = {}, acc1 = {}, acc2 = {}, acc3 = {};

    // staging assignments (128 threads)
    const int arow = tid >> 1;              // 0..63
    const int akk  = (tid & 1) * 16;        // 0 or 16
    const int bkp  = tid & 15;              // k-pair 0..15 (k = 2*bkp)
    const int bn0  = (tid >> 4) * 8;        // 0..56

    for (int k0 = 0; k0 < K; k0 += 32) {
        {   // A tile 64x32: float4 loads, 2x uint4 packed stores
            const float* src = A + (size_t)(mBase + arow) * K + k0 + akk;
            const float4 x0 = *(const float4*)(src);
            const float4 x1 = *(const float4*)(src + 4);
            const float4 x2 = *(const float4*)(src + 8);
            const float4 x3 = *(const float4*)(src + 12);
            uint4 w0, w1;
            w0.x = packbf(x0.x, x0.y); w0.y = packbf(x0.z, x0.w);
            w0.z = packbf(x1.x, x1.y); w0.w = packbf(x1.z, x1.w);
            w1.x = packbf(x2.x, x2.y); w1.y = packbf(x2.z, x2.w);
            w1.z = packbf(x3.x, x3.y); w1.w = packbf(x3.z, x3.w);
            unsigned int* d = &sA[arow * 16 + (akk >> 1)];
            *(uint4*)(d)     = w0;
            *(uint4*)(d + 4) = w1;
        }
        {   // W tile 32x64 -> transposed [col][kpair]
            const float* r0 = W + (size_t)(k0 + 2 * bkp) * N + nBase + bn0;
            const float* r1 = r0 + N;
            const float4 a0 = *(const float4*)(r0);
            const float4 a1 = *(const float4*)(r0 + 4);
            const float4 b0 = *(const float4*)(r1);
            const float4 b1 = *(const float4*)(r1 + 4);
            sBt[(bn0 + 0) * 16 + bkp] = packbf(a0.x, b0.x);
            sBt[(bn0 + 1) * 16 + bkp] = packbf(a0.y, b0.y);
            sBt[(bn0 + 2) * 16 + bkp] = packbf(a0.z, b0.z);
            sBt[(bn0 + 3) * 16 + bkp] = packbf(a0.w, b0.w);
            sBt[(bn0 + 4) * 16 + bkp] = packbf(a1.x, b1.x);
            sBt[(bn0 + 5) * 16 + bkp] = packbf(a1.y, b1.y);
            sBt[(bn0 + 6) * 16 + bkp] = packbf(a1.z, b1.z);
            sBt[(bn0 + 7) * 16 + bkp] = packbf(a1.w, b1.w);
        }
        __syncthreads();

        const unsigned short* sa = (const unsigned short*)sA;
        const unsigned short* sb = (const unsigned short*)sBt;
        const v16bf af  = load_frag(&sa[(wave * 16 + l16) * 32 + hi * 8]);
        const v16bf bf0 = load_frag(&sb[( 0 + l16) * 32 + hi * 8]);
        const v16bf bf1 = load_frag(&sb[(16 + l16) * 32 + hi * 8]);
        const v16bf bf2 = load_frag(&sb[(32 + l16) * 32 + hi * 8]);
        const v16bf bf3 = load_frag(&sb[(48 + l16) * 32 + hi * 8]);
        acc0 = wmma_bf16(af, bf0, acc0);
        acc1 = wmma_bf16(af, bf1, acc1);
        acc2 = wmma_bf16(af, bf2, acc2);
        acc3 = wmma_bf16(af, bf3, acc3);
        __syncthreads();
    }

    // branch-free epilogue; one row-base add per row
    const v8f accs[4] = { acc0, acc1, acc2, acc3 };
    float bcol[4];
#pragma unroll
    for (int t = 0; t < 4; ++t) bcol[t] = bias[nBase + t * 16 + l16];

    const int row0 = mBase + wave * 16 + hi * 8;
    size_t base  = (size_t)row0 * N + nBase + l16;
#pragma unroll
    for (int r = 0; r < 8; ++r) {
        float* crow = C + base;
        const float* rrow = RESID ? (resid + base) : nullptr;
#pragma unroll
        for (int t = 0; t < 4; ++t) {
            float v = accs[t][r] + bcol[t];
            if (RESID) v += rrow[t * 16];
            if (RELU)  v = fmaxf(v, 0.f);
            crow[t * 16] = v;
        }
        base += N;
    }
}

// ---------------------------------------------------------------------------
// Flash attention: one block per (b,h); each wave owns 16 query rows.
// K/V staged per 32-key chunk in LDS (V transposed, packed); online softmax.
// Q,K,V,O are [B, L, D] fp32 with head slice at h*DK. Lk % 32 == 0.
// ---------------------------------------------------------------------------
__global__ void attn_kernel(const float* __restrict__ Q, const float* __restrict__ Kp,
                            const float* __restrict__ Vp, const int* __restrict__ mask,
                            float* __restrict__ O, int Lq, int Lk, float scale)
{
    __shared__ __align__(16) unsigned int  sK32 [32 * 32];  // [key][dimpair]
    __shared__ __align__(16) unsigned int  sVt32[64 * 16];  // [dim][keypair]
    __shared__ __align__(16) unsigned short sP  [8 * 16 * 32]; // per-wave P

    const int b    = blockIdx.x / Hh;
    const int h    = blockIdx.x % Hh;
    const int tid  = threadIdx.x;
    const int lane = tid & 31;
    const int wave = tid >> 5;
    const int l16  = lane & 15;
    const int hi   = lane >> 4;
    const int qbase = wave * 16;
    const int nthr  = blockDim.x;
    unsigned short* sPw = sP + wave * 16 * 32;

    // ---- Q fragments (DK=64 -> two 16x32 A-frags); pad rows >= Lq with zero
    union FU { unsigned short us[16]; v16bf bf; } qa0, qa1;
    {
        const int q  = qbase + l16;
        const bool ok = q < Lq;
        const float* qrow = Q + (size_t)(b * Lq + (ok ? q : 0)) * Dm + h * DK;
#pragma unroll
        for (int i = 0; i < 8; ++i) {
            const int k = ((i & 4) << 2) + ((i & 3) << 1) + hi * 8;
            qa0.us[2*i]   = f2bf(ok ? qrow[k]      : 0.f);
            qa0.us[2*i+1] = f2bf(ok ? qrow[k + 1]  : 0.f);
            qa1.us[2*i]   = f2bf(ok ? qrow[32 + k]     : 0.f);
            qa1.us[2*i+1] = f2bf(ok ? qrow[32 + k + 1] : 0.f);
        }
    }

    float mrow[8], lrow[8];
#pragma unroll
    for (int r = 0; r < 8; ++r) { mrow[r] = -1e30f; lrow[r] = 0.f; }
    v8f o0 = {}, o1 = {}, o2 = {}, o3 = {};

    for (int kb = 0; kb < Lk; kb += 32) {
        // K chunk [32 keys][64 dims], packed along dim (uint2 -> ds_store_b64)
        for (int e = tid; e < 512; e += nthr) {
            const int key = e >> 4, db = (e & 15) * 4;
            const float4 x = *(const float4*)(Kp +
                (size_t)(b * Lk + kb + key) * Dm + h * DK + db);
            uint2 w;
            w.x = packbf(x.x, x.y);
            w.y = packbf(x.z, x.w);
            *(uint2*)&sK32[key * 32 + (db >> 1)] = w;
        }
        // V chunk transposed [64 dims][32 keys], packed along key
        for (int e = tid; e < 256; e += nthr) {
            const int kp = e >> 4, db = (e & 15) * 4;
            const float* r0 = Vp + (size_t)(b * Lk + kb + 2 * kp) * Dm + h * DK + db;
            const float4 x0 = *(const float4*)(r0);
            const float4 x1 = *(const float4*)(r0 + Dm);
            sVt32[(db + 0) * 16 + kp] = packbf(x0.x, x1.x);
            sVt32[(db + 1) * 16 + kp] = packbf(x0.y, x1.y);
            sVt32[(db + 2) * 16 + kp] = packbf(x0.z, x1.z);
            sVt32[(db + 3) * 16 + kp] = packbf(x0.w, x1.w);
        }
        __syncthreads();

        const unsigned short* sk = (const unsigned short*)sK32;
        const unsigned short* sv = (const unsigned short*)sVt32;

        // S = Q @ K^T for two 16-key tiles (preload all K frags)
        const v16bf kf0 = load_frag(&sk[(l16) * 64 +      hi * 8]);
        const v16bf kf1 = load_frag(&sk[(l16) * 64 + 32 + hi * 8]);
        const v16bf kf2 = load_frag(&sk[(16 + l16) * 64 +      hi * 8]);
        const v16bf kf3 = load_frag(&sk[(16 + l16) * 64 + 32 + hi * 8]);
        v8f s0 = {}, s1 = {};
        s0 = wmma_bf16(qa0.bf, kf0, s0);
        s0 = wmma_bf16(qa1.bf, kf1, s0);
        s1 = wmma_bf16(qa0.bf, kf2, s1);
        s1 = wmma_bf16(qa1.bf, kf3, s1);

        const bool ok0 = mask[b * Lk + kb + l16]      != 0;
        const bool ok1 = mask[b * Lk + kb + 16 + l16] != 0;

        // online softmax update, per row held by this lane-group
#pragma unroll
        for (int r = 0; r < 8; ++r) {
            const float v0 = ok0 ? s0[r] * scale : -1e9f;
            const float v1 = ok1 ? s1[r] * scale : -1e9f;
            float tm = redmax16(fmaxf(v0, v1));
            const float mn = fmaxf(mrow[r], tm);
            const float sc = __expf(mrow[r] - mn);
            const float p0 = __expf(v0 - mn);
            const float p1 = __expf(v1 - mn);
            lrow[r] = lrow[r] * sc + redsum16(p0 + p1);
            mrow[r] = mn;
            o0[r] *= sc; o1[r] *= sc; o2[r] *= sc; o3[r] *= sc;
            const int prow = hi ? r + 8 : r;
            sPw[prow * 32 + l16]      = f2bf(p0);
            sPw[prow * 32 + 16 + l16] = f2bf(p1);
        }

        // O += P @ V   (in-wave LDS ordering guarantees P visibility)
        {
            const v16bf pa  = load_frag(&sPw[l16 * 32 + hi * 8]);
            const v16bf vf0 = load_frag(&sv[( 0 + l16) * 32 + hi * 8]);
            const v16bf vf1 = load_frag(&sv[(16 + l16) * 32 + hi * 8]);
            const v16bf vf2 = load_frag(&sv[(32 + l16) * 32 + hi * 8]);
            const v16bf vf3 = load_frag(&sv[(48 + l16) * 32 + hi * 8]);
            o0 = wmma_bf16(pa, vf0, o0);
            o1 = wmma_bf16(pa, vf1, o1);
            o2 = wmma_bf16(pa, vf2, o2);
            o3 = wmma_bf16(pa, vf3, o3);
        }
        __syncthreads();
    }

    // output epilogue: one row-base add per row
    size_t base = (size_t)(b * Lq + qbase + hi * 8) * Dm + h * DK + l16;
#pragma unroll
    for (int r = 0; r < 8; ++r) {
        const int q = qbase + hi * 8 + r;
        if (q < Lq) {
            const float inv = 1.0f / lrow[r];
            float* orow = O + base;
            orow[ 0] = o0[r] * inv;
            orow[16] = o1[r] * inv;
            orow[32] = o2[r] * inv;
            orow[48] = o3[r] * inv;
        }
        base += Dm;
    }
}

// ---------------------------------------------------------------------------
// LayerNorm over D=768 (one block of 256 threads per row), in-place capable.
// ---------------------------------------------------------------------------
__global__ void __launch_bounds__(256)
ln_kernel(const float* __restrict__ in, float* __restrict__ out,
          const float* __restrict__ g, const float* __restrict__ beta)
{
    const int row = blockIdx.x;
    const int tid = threadIdx.x;
    const float* x = in + (size_t)row * Dm;
    const float v0 = x[tid], v1 = x[tid + 256], v2 = x[tid + 512];
    float s = v0 + v1 + v2;
    float q = v0 * v0 + v1 * v1 + v2 * v2;
#pragma unroll
    for (int m = 16; m >= 1; m >>= 1) { s += __shfl_xor(s, m, 32); q += __shfl_xor(q, m, 32); }
    __shared__ float rs[8], rq[8];
    const int wave = tid >> 5, lane = tid & 31;
    if (lane == 0) { rs[wave] = s; rq[wave] = q; }
    __syncthreads();
    float S = 0.f, Qq = 0.f;
#pragma unroll
    for (int i = 0; i < 8; ++i) { S += rs[i]; Qq += rq[i]; }
    const float mean = S * (1.0f / Dm);
    const float var  = Qq * (1.0f / Dm) - mean * mean;
    const float rstd = rsqrtf(var + 1e-5f);
    float* y = out + (size_t)row * Dm;
    y[tid]       = (v0 - mean) * rstd * g[tid]       + beta[tid];
    y[tid + 256] = (v1 - mean) * rstd * g[tid + 256] + beta[tid + 256];
    y[tid + 512] = (v2 - mean) * rstd * g[tid + 512] + beta[tid + 512];
}

// ---------------------------------------------------------------------------
// [S,B,D] -> [B,S,D] permute, and va = feaSlots[S,B,D] + att[B,S,D]
// ---------------------------------------------------------------------------
__global__ void permute_sbd_bsd(const float* __restrict__ in, float* __restrict__ out)
{
    const size_t total = (size_t)Sn * Bn * Dm;
    size_t i = (size_t)blockIdx.x * blockDim.x + threadIdx.x;
    if (i >= total) return;
    const int d = (int)(i % Dm);
    const size_t t = i / Dm;
    const int b = (int)(t % Bn);
    const int s = (int)(t / Bn);
    out[(size_t)(b * Sn + s) * Dm + d] = in[i];
}

__global__ void add_perm_kernel(const float* __restrict__ base,  // [S,B,D]
                                const float* __restrict__ att,   // [B,S,D]
                                float* __restrict__ out)         // [S,B,D]
{
    const size_t total = (size_t)Sn * Bn * Dm;
    size_t i = (size_t)blockIdx.x * blockDim.x + threadIdx.x;
    if (i >= total) return;
    const int d = (int)(i % Dm);
    const size_t t = i / Dm;
    const int b = (int)(t % Bn);
    const int s = (int)(t / Bn);
    out[i] = base[i] + att[(size_t)(b * Sn + s) * Dm + d];
}

// ---------------------------------------------------------------------------
// Host orchestration
// ---------------------------------------------------------------------------
extern "C" void kernel_launch(void* const* d_in, const int* in_sizes, int n_in,
                              void* d_out, int out_size, void* d_ws, size_t ws_size,
                              hipStream_t stream)
{
    (void)in_sizes; (void)n_in; (void)out_size; (void)ws_size;

    const float* cur_raw   = (const float*)d_in[0];   // [B,LU,DIN]
    const float* ctx_raw   = (const float*)d_in[1];   // [B,LC,DIN]
    const float* slots_raw = (const float*)d_in[2];   // [S,B,DIN]
    const int*   ctx_mask  = (const int*)  d_in[3];   // [B,1,LC]
    const int*   cur_mask  = (const int*)  d_in[4];   // [B,1,LU]
    const float* norm_W    = (const float*)d_in[5];   // [3,DIN,D]
    const float* norm_b    = (const float*)d_in[6];
    const float* norm_g    = (const float*)d_in[7];
    const float* norm_beta = (const float*)d_in[8];
    const float* Wq  = (const float*)d_in[9];
    const float* bq  = (const float*)d_in[10];
    const float* Wk  = (const float*)d_in[11];
    const float* bk  = (const float*)d_in[12];
    const float* Wv  = (const float*)d_in[13];
    const float* bv  = (const float*)d_in[14];
    const float* Wo  = (const float*)d_in[15];
    const float* bo  = (const float*)d_in[16];
    const float* lng = (const float*)d_in[17];
    const float* lnb = (const float*)d_in[18];
    const float* W1  = (const float*)d_in[19];
    const float* b1  = (const float*)d_in[20];
    const float* W2  = (const float*)d_in[21];
    const float* b2  = (const float*)d_in[22];

    const size_t NC = (size_t)Bn * LU * Dm;    // 3,145,728
    const size_t NX = (size_t)Bn * LC * Dm;    // 12,582,912
    const size_t NS = (size_t)Sn * Bn * Dm;    //   737,280
    const size_t DD = (size_t)Dm * Dm;

    float* ws = (float*)d_ws;
    float* feaCtx   = ws;                       // NX (phase2: reused below)
    float* K0       = feaCtx + NX;              // NX (phase2: K1)
    float* V0       = K0 + NX;                  // NX (phase2: V1)
    float* feaCur   = V0 + NX;                  // NC
    float* Q0       = feaCur + NC;              // NC
    float* att0     = Q0 + NC;                  // NC
    float* curAttn  = att0 + NC;                // NC
    float* ln0      = curAttn + NC;             // NC
    float* ffnh0    = ln0 + NC;                 // B*LU*DFF
    float* feaSlots = ffnh0 + (size_t)Bn * LU * DFF;   // NS
    float* slotsBS  = feaSlots + NS;            // NS
    float* Q1       = slotsBS + NS;             // NS
    // phase-2 reuse
    float* K1      = K0;
    float* V1      = V0;
    float* att1    = feaCtx;                    // NS
    float* slotAtt = feaCtx + NS;               // NS
    float* va      = slotAtt + NS;              // NS
    float* ln1     = va + NS;                   // NS
    float* ffnh1   = ln1 + NS;                  // S*B*DFF

    float* curOut   = (float*)d_out;            // [B,LU,D]
    float* slotsOut = curOut + NC;              // [S,B,D]

    const dim3 blkG(128);
    const dim3 blk(256);
    auto gg = [](int M, int N) { return dim3(N / 64, M / 64); };

    // ---- input projections + LayerNorm (in-place LN) --------------------
    gemm_kernel<false,false><<<gg(Bn*LU, Dm), blkG, 0, stream>>>(cur_raw,   norm_W + 0*DD, norm_b + 0*Dm, nullptr, feaCur,   Bn*LU, Dm, Dm);
    gemm_kernel<false,false><<<gg(Bn*LC, Dm), blkG, 0, stream>>>(ctx_raw,   norm_W + 1*DD, norm_b + 1*Dm, nullptr, feaCtx,   Bn*LC, Dm, Dm);
    gemm_kernel<false,false><<<gg(Sn*Bn, Dm), blkG, 0, stream>>>(slots_raw, norm_W + 2*DD, norm_b + 2*Dm, nullptr, feaSlots, Sn*Bn, Dm, Dm);
    ln_kernel<<<Bn*LU, blk, 0, stream>>>(feaCur,   feaCur,   norm_g + 0*Dm, norm_beta + 0*Dm);
    ln_kernel<<<Bn*LC, blk, 0, stream>>>(feaCtx,   feaCtx,   norm_g + 1*Dm, norm_beta + 1*Dm);
    ln_kernel<<<Sn*Bn, blk, 0, stream>>>(feaSlots, feaSlots, norm_g + 2*Dm, norm_beta + 2*Dm);

    // ---- phase 1: cur_utt cross-attends to context ----------------------
    gemm_kernel<false,false><<<gg(Bn*LU, Dm), blkG, 0, stream>>>(feaCur, Wq + 0*DD, bq + 0*Dm, nullptr, Q0, Bn*LU, Dm, Dm);
    gemm_kernel<false,false><<<gg(Bn*LC, Dm), blkG, 0, stream>>>(feaCtx, Wk + 0*DD, bk + 0*Dm, nullptr, K0, Bn*LC, Dm, Dm);
    gemm_kernel<false,false><<<gg(Bn*LC, Dm), blkG, 0, stream>>>(feaCtx, Wv + 0*DD, bv + 0*Dm, nullptr, V0, Bn*LC, Dm, Dm);
    attn_kernel<<<dim3(Bn*Hh), dim3(256), 0, stream>>>(Q0, K0, V0, ctx_mask, att0, LU, LC, 0.125f);
    // cur_attn = fea_cur + att0 @ Wo + bo
    gemm_kernel<false,true><<<gg(Bn*LU, Dm), blkG, 0, stream>>>(att0, Wo + 0*DD, bo + 0*Dm, feaCur, curAttn, Bn*LU, Dm, Dm);
    ln_kernel<<<Bn*LU, blk, 0, stream>>>(curAttn, ln0, lng + 0*Dm, lnb + 0*Dm);
    gemm_kernel<true,false><<<gg(Bn*LU, DFF), blkG, 0, stream>>>(ln0,   W1 + 0*(size_t)Dm*DFF, b1 + 0*DFF, nullptr, ffnh0, Bn*LU, DFF, Dm);
    gemm_kernel<false,true><<<gg(Bn*LU, Dm),  blkG, 0, stream>>>(ffnh0, W2 + 0*(size_t)DFF*Dm, b2 + 0*Dm,  curAttn, curOut, Bn*LU, Dm, DFF);

    // ---- phase 2: slots attend to cur_out -------------------------------
    const int nPerm = (int)(((size_t)Sn * Bn * Dm + 255) / 256);
    permute_sbd_bsd<<<nPerm, blk, 0, stream>>>(feaSlots, slotsBS);
    gemm_kernel<false,false><<<gg(Sn*Bn, Dm), blkG, 0, stream>>>(slotsBS, Wq + 1*DD, bq + 1*Dm, nullptr, Q1, Sn*Bn, Dm, Dm);
    gemm_kernel<false,false><<<gg(Bn*LU, Dm), blkG, 0, stream>>>(curOut,  Wk + 1*DD, bk + 1*Dm, nullptr, K1, Bn*LU, Dm, Dm);
    gemm_kernel<false,false><<<gg(Bn*LU, Dm), blkG, 0, stream>>>(curOut,  Wv + 1*DD, bv + 1*Dm, nullptr, V1, Bn*LU, Dm, Dm);
    attn_kernel<<<dim3(Bn*Hh), dim3(64), 0, stream>>>(Q1, K1, V1, cur_mask, att1, Sn, LU, 0.125f);
    gemm_kernel<false,false><<<gg(Sn*Bn, Dm), blkG, 0, stream>>>(att1, Wo + 1*DD, bo + 1*Dm, nullptr, slotAtt, Sn*Bn, Dm, Dm);
    add_perm_kernel<<<nPerm, blk, 0, stream>>>(feaSlots, slotAtt, va);
    ln_kernel<<<Sn*Bn, blk, 0, stream>>>(va, ln1, lng + 1*Dm, lnb + 1*Dm);
    gemm_kernel<true,false><<<gg(Sn*Bn, DFF), blkG, 0, stream>>>(ln1,   W1 + 1*(size_t)Dm*DFF, b1 + 1*DFF, nullptr, ffnh1,    Sn*Bn, DFF, Dm);
    gemm_kernel<false,true><<<gg(Sn*Bn, Dm),  blkG, 0, stream>>>(ffnh1, W2 + 1*(size_t)DFF*Dm, b2 + 1*Dm,  va,      slotsOut, Sn*Bn, Dm, DFF);
}
